// MultiHeadAttention_69535520522513
// MI455X (gfx1250) — matrix operand order repaired and checked by reading, not compiled, not run on hardware
//
#include <hip/hip_runtime.h>
#include <hip/hip_bf16.h>

#define DEV __device__ __forceinline__

typedef __attribute__((ext_vector_type(16))) __bf16 bf16x16;
typedef __attribute__((ext_vector_type(8)))  __bf16 bf16x8;
typedef __attribute__((ext_vector_type(8)))  float  f32x8;
typedef __attribute__((ext_vector_type(4)))  float  f32x4;

static constexpr int D_  = 1024;
static constexpr int H_  = 16;
static constexpr int HD_ = 64;
static constexpr int S_  = 2048;
static constexpr int B_  = 2;

// ---- WMMA fragment builders (layouts per CDNA5 ISA 7.12.2, wave32) -------
DEV bf16x16 make_a_frag(const __bf16* rowBase, int g) {
  bf16x8 lo = *(const bf16x8*)(rowBase + g * 8);
  bf16x8 hi = *(const bf16x8*)(rowBase + 16 + g * 8);
  return __builtin_shufflevector(lo, hi, 0,1,2,3,4,5,6,7,8,9,10,11,12,13,14,15);
}
DEV bf16x16 make_b_frag(const __bf16* colBase, int g) {
  bf16x8 lo = *(const bf16x8*)(colBase + g * 8);
  bf16x8 hi = *(const bf16x8*)(colBase + 16 + g * 8);
  return __builtin_shufflevector(lo, hi, 0,1,2,3,4,5,6,7,8,9,10,11,12,13,14,15);
}
DEV f32x8 wmma_bf16(bf16x16 a, bf16x16 b, f32x8 c) {
  return __builtin_amdgcn_wmma_f32_16x16x32_bf16(false, a, false, b,
                                                 (short)0, c, false, false);
}

// ---- CDNA5 async global->LDS copy (ASYNCcnt path) -------------------------
// AMDGPU flat->local addrspacecast is a low-32-bit truncate, so the low 32
// bits of a generic pointer into __shared__ are the LDS byte address.
DEV unsigned to_lds(const void* p) {
  return (unsigned)(unsigned long long)p;
}
template <int OFF>
DEV void async_ld_b128(unsigned ldsAddr, const void* gptr) {
  // INST_OFFSET is added to BOTH the LDS and global address (ISA 10.7/15.18.3)
  asm volatile("global_load_async_to_lds_b128 %0, %1, off offset:%2"
               :: "v"(ldsAddr),
                  "v"((unsigned long long)(size_t)gptr),
                  "i"(OFF)
               : "memory");
}
DEV void wait_async0() { asm volatile("s_wait_asynccnt 0x0" ::: "memory"); }

// ===========================================================================
// Kernel 1: fused QKV projection.  y = x * W^T + b, split to [B,H,S,Hd] bf16.
// Block 256 thr (8 waves), tile 128(M) x 64(N), BK=32, fp32->bf16 in staging.
// ===========================================================================
__global__ __launch_bounds__(256)
void qkv_kernel(const float* __restrict__ x,
                const float* __restrict__ Wq, const float* __restrict__ bq,
                const float* __restrict__ Wk, const float* __restrict__ bk,
                const float* __restrict__ Wv, const float* __restrict__ bv,
                __bf16* __restrict__ Qb, __bf16* __restrict__ Kb,
                __bf16* __restrict__ Vb)
{
  __shared__ __bf16 ldsA[128][40];   // 80B rows: 16B-aligned, bank-spread
  __shared__ __bf16 ldsW[64][40];

  const int z = blockIdx.z;
  const float* W    = (z == 0) ? Wq : (z == 1) ? Wk : Wv;
  const float* bias = (z == 0) ? bq : (z == 1) ? bk : bv;
  __bf16*      dst  = (z == 0) ? Qb : (z == 1) ? Kb : Vb;

  const int m0 = blockIdx.y * 128;
  const int n0 = blockIdx.x * 64;
  const int tid  = threadIdx.x;
  const int lane = tid & 31, wid = tid >> 5;
  const int g = lane >> 4, ln = lane & 15;
  const int waveM = (wid & 3) * 32;
  const int waveN = (wid >> 2) * 32;

  const int arow = tid >> 1, acol = (tid & 1) * 16;
  const int wrow = tid >> 2, wcol = (tid & 3) * 8;

  f32x8 c[2][2] = {};

  for (int k0 = 0; k0 < D_; k0 += 32) {
    __syncthreads();
    {   // stage A: x fp32 -> bf16
      const float* src = x + (size_t)(m0 + arow) * D_ + k0 + acol;
      #pragma unroll
      for (int i = 0; i < 16; i += 4) {
        f32x4 v = *(const f32x4*)(src + i);
        ldsA[arow][acol + i + 0] = (__bf16)v.x;
        ldsA[arow][acol + i + 1] = (__bf16)v.y;
        ldsA[arow][acol + i + 2] = (__bf16)v.z;
        ldsA[arow][acol + i + 3] = (__bf16)v.w;
      }
    }
    {   // stage W row-major-in-k == B-frag layout
      const float* src = W + (size_t)(n0 + wrow) * D_ + k0 + wcol;
      f32x4 v0 = *(const f32x4*)(src);
      f32x4 v1 = *(const f32x4*)(src + 4);
      ldsW[wrow][wcol + 0] = (__bf16)v0.x;  ldsW[wrow][wcol + 1] = (__bf16)v0.y;
      ldsW[wrow][wcol + 2] = (__bf16)v0.z;  ldsW[wrow][wcol + 3] = (__bf16)v0.w;
      ldsW[wrow][wcol + 4] = (__bf16)v1.x;  ldsW[wrow][wcol + 5] = (__bf16)v1.y;
      ldsW[wrow][wcol + 6] = (__bf16)v1.z;  ldsW[wrow][wcol + 7] = (__bf16)v1.w;
    }
    if (k0 + 32 < D_) {
      __builtin_prefetch(x + (size_t)(m0 + arow) * D_ + k0 + 32 + acol, 0, 3);
      __builtin_prefetch(W + (size_t)(n0 + wrow) * D_ + k0 + 32 + wcol, 0, 3);
    }
    __syncthreads();

    bf16x16 bfr[2];
    #pragma unroll
    for (int ni = 0; ni < 2; ++ni)
      bfr[ni] = make_b_frag(&ldsW[waveN + ni * 16 + ln][0], g);
    #pragma unroll
    for (int mi = 0; mi < 2; ++mi) {
      bf16x16 a = make_a_frag(&ldsA[waveM + mi * 16 + ln][0], g);
      #pragma unroll
      for (int ni = 0; ni < 2; ++ni)
        c[mi][ni] = wmma_bf16(a, bfr[ni], c[mi][ni]);
    }
  }

  #pragma unroll
  for (int mi = 0; mi < 2; ++mi)
    #pragma unroll
    for (int ni = 0; ni < 2; ++ni)
      #pragma unroll
      for (int r = 0; r < 8; ++r) {
        int row = m0 + waveM + mi * 16 + r + g * 8;
        int col = n0 + waveN + ni * 16 + ln;
        float v = c[mi][ni][r] + bias[col];
        int bb = row >> 11, ss = row & (S_ - 1);
        int hh = col >> 6,  dd = col & (HD_ - 1);
        dst[(((size_t)(bb * H_ + hh)) * S_ + ss) * HD_ + dd] = (__bf16)v;
      }
}

// ===========================================================================
// Kernel 2: flash attention per (b, h, 64-query tile). 128 thr = 4 waves.
// K/V tiles double-buffered through global_load_async_to_lds_b128; copy of
// tile i+1 overlaps WMMA+softmax of tile i.
// ===========================================================================
__global__ __launch_bounds__(128)
void attn_kernel(const __bf16* __restrict__ Qb, const __bf16* __restrict__ Kb,
                 const __bf16* __restrict__ Vb, __bf16* __restrict__ attnb)
{
  __shared__ __bf16 ldsQ [64][72];      // 144B rows: 16B-aligned, bank-spread
  __shared__ __bf16 ldsK [2][64][72];   // async double buffer
  __shared__ __bf16 ldsVr[2][64][72];   // async double buffer (row major)
  __shared__ __bf16 ldsVT[64][72];      // V transposed: [d][key]
  __shared__ __bf16 ldsP [64][72];      // probabilities (bf16) for PV WMMA

  const int q0 = blockIdx.x * 64;
  const int h  = blockIdx.y, b = blockIdx.z;
  const int tid  = threadIdx.x;
  const int lane = tid & 31, w = tid >> 5;
  const int g = lane >> 4, ln = lane & 15;

  const size_t headBase = ((size_t)(b * H_ + h)) * S_ * HD_;
  const __bf16* Qh = Qb + headBase;
  const __bf16* Kh = Kb + headBase;
  const __bf16* Vh = Vb + headBase;

  const int srow = tid >> 1, scolh = (tid & 1) * 32;  // 64B row chunk / lane

  auto issueKV = [&](int buf, int kt) {
    const __bf16* kg = Kh + (size_t)(kt + srow) * HD_ + scolh;
    const __bf16* vg = Vh + (size_t)(kt + srow) * HD_ + scolh;
    unsigned kl = to_lds(&ldsK [buf][srow][scolh]);
    unsigned vl = to_lds(&ldsVr[buf][srow][scolh]);
    async_ld_b128<0>(kl, kg);  async_ld_b128<16>(kl, kg);
    async_ld_b128<32>(kl, kg); async_ld_b128<48>(kl, kg);
    async_ld_b128<0>(vl, vg);  async_ld_b128<16>(vl, vg);
    async_ld_b128<32>(vl, vg); async_ld_b128<48>(vl, vg);
  };

  {   // Q tile (once) via async path too
    const __bf16* qg = Qh + (size_t)(q0 + srow) * HD_ + scolh;
    unsigned ql = to_lds(&ldsQ[srow][scolh]);
    async_ld_b128<0>(ql, qg);  async_ld_b128<16>(ql, qg);
    async_ld_b128<32>(ql, qg); async_ld_b128<48>(ql, qg);
  }
  issueKV(0, 0);

  float m_i[8], l_i[8];
  f32x8 o[4] = {};
  #pragma unroll
  for (int r = 0; r < 8; ++r) { m_i[r] = -3.0e38f; l_i[r] = 0.0f; }
  const float scale = 0.125f;   // 1/sqrt(64)

  constexpr int NT = S_ / 64;
  for (int it = 0; it < NT; ++it) {
    const int cur = it & 1;
    wait_async0();           // my async copies for tile `it` (and Q) done
    __syncthreads();         // everyone's copies visible; prev buffer free

    if (it + 1 < NT) issueKV(1 - cur, (it + 1) * 64);   // overlap next copy

    // transpose V (LDS->LDS) into B-frag-friendly [d][key]
    #pragma unroll
    for (int i = 0; i < 32; ++i)
      ldsVT[scolh + i][srow] = ldsVr[cur][srow][scolh + i];

    // ---- scores S = scale * Q K^T  (16 q-rows x 64 keys per wave) --------
    bf16x16 a0 = make_a_frag(&ldsQ[w * 16 + ln][0],  g);
    bf16x16 a1 = make_a_frag(&ldsQ[w * 16 + ln][32], g);
    f32x8 s[4];
    #pragma unroll
    for (int f = 0; f < 4; ++f) {
      f32x8 acc = {};
      bf16x16 b0 = make_b_frag(&ldsK[cur][f * 16 + ln][0],  g);
      bf16x16 b1 = make_b_frag(&ldsK[cur][f * 16 + ln][32], g);
      acc = wmma_bf16(a0, b0, acc);
      acc = wmma_bf16(a1, b1, acc);
      s[f] = acc * scale;
    }

    // ---- online softmax; rows r+g*8 live in 16-lane halves ---------------
    #pragma unroll
    for (int r = 0; r < 8; ++r) {
      float mx = fmaxf(fmaxf(s[0][r], s[1][r]), fmaxf(s[2][r], s[3][r]));
      mx = fmaxf(mx, __shfl_xor(mx, 1));
      mx = fmaxf(mx, __shfl_xor(mx, 2));
      mx = fmaxf(mx, __shfl_xor(mx, 4));
      mx = fmaxf(mx, __shfl_xor(mx, 8));
      float newm = fmaxf(m_i[r], mx);
      float al   = __expf(m_i[r] - newm);
      float rs = 0.0f;
      #pragma unroll
      for (int f = 0; f < 4; ++f) {
        float p = __expf(s[f][r] - newm);
        s[f][r] = p;
        rs += p;
      }
      rs += __shfl_xor(rs, 1); rs += __shfl_xor(rs, 2);
      rs += __shfl_xor(rs, 4); rs += __shfl_xor(rs, 8);
      l_i[r] = l_i[r] * al + rs;
      m_i[r] = newm;
      #pragma unroll
      for (int f = 0; f < 4; ++f) o[f][r] *= al;
    }

    // ---- P (bf16) through LDS ---------------------------------------------
    #pragma unroll
    for (int f = 0; f < 4; ++f)
      #pragma unroll
      for (int r = 0; r < 8; ++r)
        ldsP[w * 16 + r + g * 8][f * 16 + ln] = (__bf16)s[f][r];

    __syncthreads();   // V^T + P visible to all waves

    // ---- O += P V ---------------------------------------------------------
    bf16x16 p0 = make_a_frag(&ldsP[w * 16 + ln][0],  g);
    bf16x16 p1 = make_a_frag(&ldsP[w * 16 + ln][32], g);
    #pragma unroll
    for (int f = 0; f < 4; ++f) {
      bf16x16 v0 = make_b_frag(&ldsVT[f * 16 + ln][0],  g);
      bf16x16 v1 = make_b_frag(&ldsVT[f * 16 + ln][32], g);
      o[f] = wmma_bf16(p0, v0, o[f]);
      o[f] = wmma_bf16(p1, v1, o[f]);
    }
  }

  // normalize and store to [B,S,D] bf16 (merged heads)
  #pragma unroll
  for (int r = 0; r < 8; ++r) {
    float inv = 1.0f / l_i[r];
    int row = b * S_ + q0 + w * 16 + r + g * 8;
    #pragma unroll
    for (int f = 0; f < 4; ++f) {
      int col = h * HD_ + f * 16 + ln;
      attnb[(size_t)row * D_ + col] = (__bf16)(o[f][r] * inv);
    }
  }
}

// ===========================================================================
// Kernel 3: out = attn * Wo^T + bo.  A tile (bf16) staged via async copies.
// ===========================================================================
__global__ __launch_bounds__(256)
void oproj_kernel(const __bf16* __restrict__ Ab, const float* __restrict__ Wo,
                  const float* __restrict__ bo, float* __restrict__ out)
{
  __shared__ __bf16 ldsA[128][40];
  __shared__ __bf16 ldsW[64][40];

  const int m0 = blockIdx.y * 128;
  const int n0 = blockIdx.x * 64;
  const int tid  = threadIdx.x;
  const int lane = tid & 31, wid = tid >> 5;
  const int g = lane >> 4, ln = lane & 15;
  const int waveM = (wid & 3) * 32;
  const int waveN = (wid >> 2) * 32;

  const int arow = tid >> 1, acol = (tid & 1) * 16;
  const int wrow = tid >> 2, wcol = (tid & 3) * 8;

  f32x8 c[2][2] = {};

  for (int k0 = 0; k0 < D_; k0 += 32) {
    __syncthreads();
    {   // stage A (bf16): async global->LDS, 32B per thread
      unsigned al = to_lds(&ldsA[arow][acol]);
      const __bf16* ag = Ab + (size_t)(m0 + arow) * D_ + k0 + acol;
      async_ld_b128<0>(al, ag);
      async_ld_b128<16>(al, ag);
    }
    {   // stage Wo fp32 -> bf16
      const float* src = Wo + (size_t)(n0 + wrow) * D_ + k0 + wcol;
      f32x4 v0 = *(const f32x4*)(src);
      f32x4 v1 = *(const f32x4*)(src + 4);
      ldsW[wrow][wcol + 0] = (__bf16)v0.x;  ldsW[wrow][wcol + 1] = (__bf16)v0.y;
      ldsW[wrow][wcol + 2] = (__bf16)v0.z;  ldsW[wrow][wcol + 3] = (__bf16)v0.w;
      ldsW[wrow][wcol + 4] = (__bf16)v1.x;  ldsW[wrow][wcol + 5] = (__bf16)v1.y;
      ldsW[wrow][wcol + 6] = (__bf16)v1.z;  ldsW[wrow][wcol + 7] = (__bf16)v1.w;
    }
    if (k0 + 32 < D_)
      __builtin_prefetch(Wo + (size_t)(n0 + wrow) * D_ + k0 + 32 + wcol, 0, 3);
    wait_async0();
    __syncthreads();

    bf16x16 bfr[2];
    #pragma unroll
    for (int ni = 0; ni < 2; ++ni)
      bfr[ni] = make_b_frag(&ldsW[waveN + ni * 16 + ln][0], g);
    #pragma unroll
    for (int mi = 0; mi < 2; ++mi) {
      bf16x16 a = make_a_frag(&ldsA[waveM + mi * 16 + ln][0], g);
      #pragma unroll
      for (int ni = 0; ni < 2; ++ni)
        c[mi][ni] = wmma_bf16(a, bfr[ni], c[mi][ni]);
    }
  }

  #pragma unroll
  for (int mi = 0; mi < 2; ++mi)
    #pragma unroll
    for (int ni = 0; ni < 2; ++ni)
      #pragma unroll
      for (int r = 0; r < 8; ++r) {
        int row = m0 + waveM + mi * 16 + r + g * 8;
        int col = n0 + waveN + ni * 16 + ln;
        out[(size_t)row * D_ + col] = c[mi][ni][r] + bo[col];
      }
}

// ===========================================================================
extern "C" void kernel_launch(void* const* d_in, const int* in_sizes, int n_in,
                              void* d_out, int out_size, void* d_ws,
                              size_t ws_size, hipStream_t stream)
{
  (void)in_sizes; (void)n_in; (void)out_size; (void)ws_size;
  const float* x  = (const float*)d_in[0];
  const float* Wq = (const float*)d_in[1];
  const float* bq = (const float*)d_in[2];
  const float* Wk = (const float*)d_in[3];
  const float* bk = (const float*)d_in[4];
  const float* Wv = (const float*)d_in[5];
  const float* bv = (const float*)d_in[6];
  const float* Wo = (const float*)d_in[7];
  const float* bo = (const float*)d_in[8];
  float* out = (float*)d_out;

  // workspace: Q, K, V ([B,H,S,Hd] bf16) + attn ([B,S,D] bf16) = 32 MiB
  char* ws = (char*)d_ws;
  const size_t qkvBytes = (size_t)B_ * S_ * D_ * sizeof(__bf16);   // 8 MiB
  __bf16* Qb = (__bf16*)(ws + 0 * qkvBytes);
  __bf16* Kb = (__bf16*)(ws + 1 * qkvBytes);
  __bf16* Vb = (__bf16*)(ws + 2 * qkvBytes);
  __bf16* Ab = (__bf16*)(ws + 3 * qkvBytes);

  qkv_kernel<<<dim3(D_ / 64, (B_ * S_) / 128, 3), 256, 0, stream>>>(
      x, Wq, bq, Wk, bk, Wv, bv, Qb, Kb, Vb);
  attn_kernel<<<dim3(S_ / 64, H_, B_), 128, 0, stream>>>(Qb, Kb, Vb, Ab);
  oproj_kernel<<<dim3(D_ / 64, (B_ * S_) / 128, 1), 256, 0, stream>>>(
      Ab, Wo, bo, out);
}
